// GCLMessage_35150012351069
// MI455X (gfx1250) — matrix operand order repaired and checked by reading, not compiled, not run on hardware
//
#include <hip/hip_runtime.h>
#include <hip/hip_bf16.h>

typedef __attribute__((ext_vector_type(16))) _Float16 v16h;
typedef __attribute__((ext_vector_type(8)))  _Float16 v8h;
typedef __attribute__((ext_vector_type(8)))  float    v8f;

#define HID   128
#define EDIM  416
#define K1    672   // 2*HID + EDIM
#define KN    256   // 2*HID

__device__ __forceinline__ float siluf(float x) {
    return x * (1.0f / (1.0f + __expf(-x)));
}

__device__ __forceinline__ v8f wmma16(v16h a, v16h b, v8f c) {
    return __builtin_amdgcn_wmma_f32_16x16x32_f16(
        /*neg_a=*/false, a, /*neg_b=*/false, b,
        /*c_mod=*/(short)0, c, /*reuse_a=*/false, /*reuse_b=*/false);
}

// Load a 16x32 f16 A-fragment for one lane.
// Per ISA layout: lane l (row = l&15, kbase = (l>>4)*8) holds
// K = {kb..kb+7} in elements 0..7 and K = {kb+16..kb+23} in elements 8..15.
__device__ __forceinline__ v16h load_a_frag(const _Float16* base, int stride,
                                            int row, int kb) {
    const v8h lo = *(const v8h*)(base + (size_t)row * stride + kb);
    const v8h hi = *(const v8h*)(base + (size_t)row * stride + kb + 16);
    v16h r;
#pragma unroll
    for (int t = 0; t < 8; ++t) { r[t] = lo[t]; r[8 + t] = hi[t]; }
    return r;
}

// ---------------------------------------------------------------------------
// Pack a row-major (K x N) f32 weight into per-lane WMMA-B fragments (f16):
//   frag f = (kt*NT + nt)*32 + lane ; elem e -> W[kt*32+(lane>>4)*8+(e&7)+(e>>3)*16][nt*16+(lane&15)]
// ---------------------------------------------------------------------------
__global__ __launch_bounds__(256) void k_pack(const float* __restrict__ W,
                                              _Float16* __restrict__ pack,
                                              int K, int N) {
    int p = blockIdx.x * 256 + threadIdx.x;
    if (p >= K * N) return;
    int e    = p & 15;
    int lane = (p >> 4) & 31;
    int pair = p >> 9;
    int NT   = N >> 4;
    int nt   = pair % NT;
    int kt   = pair / NT;
    int Kidx = kt * 32 + (lane >> 4) * 8 + (e & 7) + (e >> 3) * 16;
    int col  = nt * 16 + (lane & 15);
    pack[p] = (_Float16)W[(size_t)Kidx * N + col];
}

// ---------------------------------------------------------------------------
// LayerNorm: xh (f32) + xh_h (f16); also zero agg and cnt.
// ---------------------------------------------------------------------------
__global__ __launch_bounds__(HID) void k_layernorm(
    const float* __restrict__ x, const float* __restrict__ g,
    const float* __restrict__ b, float* __restrict__ xh,
    _Float16* __restrict__ xh_h, float* __restrict__ agg,
    float* __restrict__ cnt) {
    int i = blockIdx.x, c = threadIdx.x;
    float v = x[(size_t)i * HID + c];
    __shared__ float red[HID];
    red[c] = v;
    __syncthreads();
#pragma unroll
    for (int s = HID / 2; s > 0; s >>= 1) {
        if (c < s) red[c] += red[c + s];
        __syncthreads();
    }
    float mu = red[0] * (1.0f / HID);
    __syncthreads();
    float d = v - mu;
    red[c] = d * d;
    __syncthreads();
#pragma unroll
    for (int s = HID / 2; s > 0; s >>= 1) {
        if (c < s) red[c] += red[c + s];
        __syncthreads();
    }
    float var = red[0] * (1.0f / HID);
    float o = d * rsqrtf(var + 1e-5f) * g[c] + b[c];
    xh[(size_t)i * HID + c] = o;
    xh_h[(size_t)i * HID + c] = (_Float16)o;
    agg[(size_t)i * HID + c] = 0.0f;
    if (c == 0) cnt[i] = 0.0f;
}

// ---------------------------------------------------------------------------
// Edge kernel: one wave handles 32 edges = two 16-row M-tiles, so every
// packed-B fragment load feeds TWO v_wmma ops (halves B-fetch per FLOP).
//   cat -> silu(cat@We1+b) -> silu(@We2+b) -> att -> m_ij
//   -> atomic segment-sum into agg/cnt -> edge_out = weight + silu(m_ij@Wo+bo)
// ---------------------------------------------------------------------------
__global__ __launch_bounds__(32) void k_edge(
    const int* __restrict__ ii, const int* __restrict__ jj,
    const float* __restrict__ weight, const _Float16* __restrict__ xh_h,
    const v16h* __restrict__ pWe1, const float* __restrict__ be1,
    const v16h* __restrict__ pWe2, const float* __restrict__ be2,
    const float* __restrict__ Wa, const float* __restrict__ ba,
    const v16h* __restrict__ pWo, const float* __restrict__ bo,
    float* __restrict__ agg, float* __restrict__ cnt,
    float* __restrict__ outE) {
    // LDS: catH [32][672] f16 (43008B) ; m1H [32][128] f16 (8192B) ; att[32]
    // m2F [32][128] f32 (16384B) and mijH [32][128] f16 (8192B) overlay catH.
    __shared__ __align__(16) unsigned char smem[32 * K1 * 2 + 32 * HID * 2 + 128];
    _Float16* catH = (_Float16*)smem;
    _Float16* m1H  = (_Float16*)(smem + 32 * K1 * 2);
    float*    attS = (float*)(smem + 32 * K1 * 2 + 32 * HID * 2);
    float*    m2F  = (float*)smem;                       // 16384 B
    _Float16* mijH = (_Float16*)(smem + 32 * HID * 4);   // +8192 B (< catH)

    const int lane = threadIdx.x;
    const int e0   = blockIdx.x * 32;
    const int e    = e0 + lane;          // one staged edge per lane
    const int ni   = ii[e];
    const int nj   = jj[e];
    // WMMA fragment roles
    const int er      = lane & 15;
    const int part    = lane >> 4;
    const int kb_lane = part * 8;
    const int mhalf   = part * 8;
    const int ncol    = er;

    // ---- stage cat row: one lane per edge ----
    {
        const uint4* src = (const uint4*)(xh_h + (size_t)ni * HID);
        uint4* dst = (uint4*)(catH + (size_t)lane * K1);
#pragma unroll
        for (int t = 0; t < 16; ++t) dst[t] = src[t];
        src = (const uint4*)(xh_h + (size_t)nj * HID);
        dst = (uint4*)(catH + (size_t)lane * K1 + HID);
#pragma unroll
        for (int t = 0; t < 16; ++t) dst[t] = src[t];
        const float4* wr = (const float4*)(weight + (size_t)e * EDIM);
        _Float16* dw = catH + (size_t)lane * K1 + 2 * HID;
#pragma unroll
        for (int t = 0; t < 52; ++t) {
            float4 a = wr[2 * t], b = wr[2 * t + 1];
            v8h h;
            h[0] = (_Float16)a.x; h[1] = (_Float16)a.y;
            h[2] = (_Float16)a.z; h[3] = (_Float16)a.w;
            h[4] = (_Float16)b.x; h[5] = (_Float16)b.y;
            h[6] = (_Float16)b.z; h[7] = (_Float16)b.w;
            *(v8h*)(dw + 8 * t) = h;
        }
    }
    __syncthreads();

    // ---- layer 1: [32x672] x [672x128] ----
#pragma unroll 1
    for (int nt = 0; nt < 8; ++nt) {
        v8f acc0 = {}, acc1 = {};
#pragma unroll
        for (int kt = 0; kt < 21; ++kt) {
            int kb = kt * 32 + kb_lane;
            v16h a0 = load_a_frag(catH, K1, er, kb);
            v16h a1 = load_a_frag(catH, K1, er + 16, kb);
            v16h b  = pWe1[(kt * 8 + nt) * 32 + lane];
            acc0 = wmma16(a0, b, acc0);
            acc1 = wmma16(a1, b, acc1);
        }
        float bias = be1[nt * 16 + ncol];
#pragma unroll
        for (int v = 0; v < 8; ++v) {
            m1H[(mhalf + v) * HID + nt * 16 + ncol] =
                (_Float16)siluf(acc0[v] + bias);
            m1H[(16 + mhalf + v) * HID + nt * 16 + ncol] =
                (_Float16)siluf(acc1[v] + bias);
        }
    }
    __syncthreads();

    // ---- layer 2: [32x128] x [128x128] -> m2F (f32, overlays catH) ----
#pragma unroll 1
    for (int nt = 0; nt < 8; ++nt) {
        v8f acc0 = {}, acc1 = {};
#pragma unroll
        for (int kt = 0; kt < 4; ++kt) {
            int kb = kt * 32 + kb_lane;
            v16h a0 = load_a_frag(m1H, HID, er, kb);
            v16h a1 = load_a_frag(m1H, HID, er + 16, kb);
            v16h b  = pWe2[(kt * 8 + nt) * 32 + lane];
            acc0 = wmma16(a0, b, acc0);
            acc1 = wmma16(a1, b, acc1);
        }
        float bias = be2[nt * 16 + ncol];
#pragma unroll
        for (int v = 0; v < 8; ++v) {
            m2F[(mhalf + v) * HID + nt * 16 + ncol] = siluf(acc0[v] + bias);
            m2F[(16 + mhalf + v) * HID + nt * 16 + ncol] = siluf(acc1[v] + bias);
        }
    }
    __syncthreads();

    // ---- attention scalar: one lane per edge ----
    {
        float ps = 0.0f;
        const float* mrow = m2F + lane * HID;
#pragma unroll
        for (int k = 0; k < HID; ++k) ps += mrow[k] * Wa[k];
        attS[lane] = siluf(ps + ba[0]);
    }
    __syncthreads();

    // ---- m_ij, mean-aggregation atomics, f16 copy for Wo GEMM ----
    {
        float a_e = attS[lane];
        const float* mrow = m2F + lane * HID;
        _Float16* drow = mijH + lane * HID;
        float* arow = agg + (size_t)ni * HID;
#pragma unroll 1
        for (int k = 0; k < HID; ++k) {
            float mv = mrow[k] * a_e;
            drow[k] = (_Float16)mv;
            unsafeAtomicAdd(arow + k, mv);
        }
        unsafeAtomicAdd(cnt + ni, 1.0f);
    }
    __syncthreads();

    // ---- edge out: weight + silu(m_ij @ Wo + bo), [32x128]x[128x416] ----
#pragma unroll 1
    for (int nt = 0; nt < 26; ++nt) {
        v8f acc0 = {}, acc1 = {};
#pragma unroll
        for (int kt = 0; kt < 4; ++kt) {
            int kb = kt * 32 + kb_lane;
            v16h a0 = load_a_frag(mijH, HID, er, kb);
            v16h a1 = load_a_frag(mijH, HID, er + 16, kb);
            v16h b  = pWo[(kt * 26 + nt) * 32 + lane];
            acc0 = wmma16(a0, b, acc0);
            acc1 = wmma16(a1, b, acc1);
        }
        int col = nt * 16 + ncol;
        float bias = bo[col];
#pragma unroll
        for (int v = 0; v < 8; ++v) {
            int eM = e0 + mhalf + v;
            outE[(size_t)eM * EDIM + col] =
                weight[(size_t)eM * EDIM + col] + siluf(acc0[v] + bias);
            int eM1 = eM + 16;
            outE[(size_t)eM1 * EDIM + col] =
                weight[(size_t)eM1 * EDIM + col] + siluf(acc1[v] + bias);
        }
    }
}

// ---------------------------------------------------------------------------
// Build hH = [xh_h | agg / max(cnt,1)] in f16, [nodes][256]
// ---------------------------------------------------------------------------
__global__ __launch_bounds__(HID) void k_norm_agg(
    const _Float16* __restrict__ xh_h, const float* __restrict__ agg,
    const float* __restrict__ cnt, _Float16* __restrict__ hH) {
    int i = blockIdx.x, c = threadIdx.x;
    float cn = cnt[i];
    cn = (cn == 0.0f) ? 1.0f : cn;
    hH[(size_t)i * KN + c]       = xh_h[(size_t)i * HID + c];
    hH[(size_t)i * KN + HID + c] = (_Float16)(agg[(size_t)i * HID + c] / cn);
}

// ---------------------------------------------------------------------------
// Node kernel: one wave per 16-node tile.  xh_out = xh + mlp(hH)
// ---------------------------------------------------------------------------
__global__ __launch_bounds__(32) void k_node(
    const _Float16* __restrict__ hH,
    const v16h* __restrict__ pWn1, const float* __restrict__ bn1,
    const v16h* __restrict__ pWn2, const float* __restrict__ bn2,
    const float* __restrict__ xh, float* __restrict__ outN) {
    __shared__ __align__(16) _Float16 n1H[16 * HID];
    const int lane = threadIdx.x;
    const int t0   = blockIdx.x * 16;
    const int er   = lane & 15;
    const int part = lane >> 4;
    const int kb_lane = part * 8, mhalf = part * 8, ncol = er;

#pragma unroll 1
    for (int nt = 0; nt < 8; ++nt) {
        v8f acc = {};
#pragma unroll
        for (int kt = 0; kt < 8; ++kt) {
            v16h a = load_a_frag(hH + (size_t)t0 * KN, KN, er, kt * 32 + kb_lane);
            v16h b = pWn1[(kt * 8 + nt) * 32 + lane];
            acc = wmma16(a, b, acc);
        }
        float bias = bn1[nt * 16 + ncol];
#pragma unroll
        for (int v = 0; v < 8; ++v)
            n1H[(mhalf + v) * HID + nt * 16 + ncol] =
                (_Float16)siluf(acc[v] + bias);
    }
    __syncthreads();
#pragma unroll 1
    for (int nt = 0; nt < 8; ++nt) {
        v8f acc = {};
#pragma unroll
        for (int kt = 0; kt < 4; ++kt) {
            v16h a = load_a_frag(n1H, HID, er, kt * 32 + kb_lane);
            v16h b = pWn2[(kt * 8 + nt) * 32 + lane];
            acc = wmma16(a, b, acc);
        }
        int col = nt * 16 + ncol;
        float bias = bn2[col];
#pragma unroll
        for (int v = 0; v < 8; ++v) {
            int node = t0 + mhalf + v;
            outN[(size_t)node * HID + col] =
                xh[(size_t)node * HID + col] + siluf(acc[v] + bias);
        }
    }
}

// ---------------------------------------------------------------------------
extern "C" void kernel_launch(void* const* d_in, const int* in_sizes, int n_in,
                              void* d_out, int out_size, void* d_ws,
                              size_t ws_size, hipStream_t stream) {
    const float* x      = (const float*)d_in[0];
    const float* weight = (const float*)d_in[1];
    const float* ln_g   = (const float*)d_in[2];
    const float* ln_b   = (const float*)d_in[3];
    const float* We1    = (const float*)d_in[4];
    const float* be1    = (const float*)d_in[5];
    const float* We2    = (const float*)d_in[6];
    const float* be2    = (const float*)d_in[7];
    const float* Wa     = (const float*)d_in[8];
    const float* ba     = (const float*)d_in[9];
    const float* Wn1    = (const float*)d_in[10];
    const float* bn1    = (const float*)d_in[11];
    const float* Wn2    = (const float*)d_in[12];
    const float* bn2    = (const float*)d_in[13];
    const float* Wo     = (const float*)d_in[14];
    const float* bo     = (const float*)d_in[15];
    const int*   eidx   = (const int*)d_in[16];

    const int nodes = in_sizes[0] / HID;     // 10000
    const int edges = in_sizes[1] / EDIM;    // 160000
    const int* ii = eidx;
    const int* jj = eidx + edges;

    float* outN = (float*)d_out;
    float* outE = outN + (size_t)nodes * HID;

    // workspace carve-out
    char* ws = (char*)d_ws;
    size_t off = 0;
    auto alloc = [&](size_t bytes) {
        size_t r = off;
        off += (bytes + 255) & ~(size_t)255;
        return r;
    };
    float*    xh    = (float*)(ws + alloc((size_t)nodes * HID * 4));
    _Float16* xh_h  = (_Float16*)(ws + alloc((size_t)nodes * HID * 2));
    float*    agg   = (float*)(ws + alloc((size_t)nodes * HID * 4));
    float*    cnt   = (float*)(ws + alloc((size_t)nodes * 4));
    _Float16* hH    = (_Float16*)(ws + alloc((size_t)nodes * KN * 2));
    _Float16* pWe1  = (_Float16*)(ws + alloc((size_t)K1 * HID * 2));
    _Float16* pWe2  = (_Float16*)(ws + alloc((size_t)HID * HID * 2));
    _Float16* pWo   = (_Float16*)(ws + alloc((size_t)HID * EDIM * 2));
    _Float16* pWn1  = (_Float16*)(ws + alloc((size_t)KN * HID * 2));
    _Float16* pWn2  = (_Float16*)(ws + alloc((size_t)HID * HID * 2));

    auto gp = [](int n) { return (n + 255) / 256; };
    k_pack<<<gp(K1 * HID), 256, 0, stream>>>(We1, pWe1, K1, HID);
    k_pack<<<gp(HID * HID), 256, 0, stream>>>(We2, pWe2, HID, HID);
    k_pack<<<gp(HID * EDIM), 256, 0, stream>>>(Wo, pWo, HID, EDIM);
    k_pack<<<gp(KN * HID), 256, 0, stream>>>(Wn1, pWn1, KN, HID);
    k_pack<<<gp(HID * HID), 256, 0, stream>>>(Wn2, pWn2, HID, HID);

    k_layernorm<<<nodes, HID, 0, stream>>>(x, ln_g, ln_b, xh, xh_h, agg, cnt);

    k_edge<<<edges / 32, 32, 0, stream>>>(
        ii, jj, weight, xh_h, (const v16h*)pWe1, be1, (const v16h*)pWe2, be2,
        Wa, ba, (const v16h*)pWo, bo, agg, cnt, outE);

    k_norm_agg<<<nodes, HID, 0, stream>>>(xh_h, agg, cnt, hH);

    k_node<<<nodes / 16, 32, 0, stream>>>(
        hH, (const v16h*)pWn1, bn1, (const v16h*)pWn2, bn2, xh, outN);
}
